// AttentionMILModel_73151882985991
// MI455X (gfx1250) — matrix operand order repaired and checked by reading, not compiled, not run on hardware
//
#include <hip/hip_runtime.h>
#include <hip/hip_bf16.h>

// ---------- types ----------
typedef __attribute__((ext_vector_type(16))) __bf16 v16bf;
typedef __attribute__((ext_vector_type(8)))  float  v8f;
typedef __attribute__((ext_vector_type(4)))  unsigned u32x4;
typedef __attribute__((ext_vector_type(8)))  int    i32x8;
typedef __attribute__((ext_vector_type(4)))  int    i32x4;
typedef __attribute__((ext_vector_type(4)))  float  f32x4;

union BFrag { v16bf v; u32x4 q[2]; };   // 16 bf16 = one WMMA A/B operand
union BF8   { u32x4 q; __bf16 h[8]; };  // 8 bf16 = 16 bytes

#define N_PATCHES 262144
#define N_BAGS    32
#define BAG       (N_PATCHES / N_BAGS)   // 8192
#define IN_DIM    1024
#define F_DIM     256
#define A_DIM     128

#define APAD 40    // LDS row stride (elems) for X k-chunk tile  (80B rows, 16B aligned)
#define HPAD 264   // LDS row stride (elems) for h tile          (528B rows, 16B aligned)

#ifndef __has_builtin
#define __has_builtin(x) 0
#endif

#if __has_builtin(__builtin_amdgcn_tensor_load_to_lds) && \
    __has_builtin(__builtin_amdgcn_s_wait_tensorcnt)
#define HAVE_TDM 1
#else
#define HAVE_TDM 0
#endif

#if __has_builtin(__builtin_amdgcn_global_store_async_from_lds_b128)
#define HAVE_ASYNC_ST 1
#else
#define HAVE_ASYNC_ST 0
#endif

// address-space-qualified pointer types for the async LDS<->global builtins
typedef __attribute__((address_space(1))) i32x4 glb_i32x4;
typedef __attribute__((address_space(3))) i32x4 lds_i32x4;

__device__ __forceinline__ unsigned lds_off(const void* p) {
  return (unsigned)(unsigned long long)
         (__attribute__((address_space(3))) const void*)p;
}

__device__ __forceinline__ v8f v8f_zero() {
  v8f z;
#pragma unroll
  for (int i = 0; i < 8; ++i) z[i] = 0.f;
  return z;
}

// pack 16 fp32 -> 16 bf16 and store 32B to LDS
__device__ __forceinline__ void cvt16_store(f32x4 f0, f32x4 f1, f32x4 f2, f32x4 f3,
                                            __bf16* d) {
  BF8 o0, o1;
#pragma unroll
  for (int i = 0; i < 4; ++i) { o0.h[i] = (__bf16)f0[i]; o0.h[4 + i] = (__bf16)f1[i]; }
#pragma unroll
  for (int i = 0; i < 4; ++i) { o1.h[i] = (__bf16)f2[i]; o1.h[4 + i] = (__bf16)f3[i]; }
  *(u32x4*)d       = o0.q;
  *(u32x4*)(d + 8) = o1.q;
}

#if HAVE_TDM
// TDM: DMA a 256-row x 32-col bf16 tile of W1 (row stride 1024 elems) into LDS.
// D# packed per CDNA5 ISA (group0 + group1; groups 2/3 zero).
__device__ __forceinline__ void tdm_load_w_chunk(const __bf16* gsrc, unsigned ldsByteOff) {
  unsigned long long ga = (unsigned long long)gsrc;
  u32x4 g0;
  g0[0] = 1u;                                       // count=1 (valid), user mode
  g0[1] = ldsByteOff;                               // lds_addr (bytes)
  g0[2] = (unsigned)(ga & 0xffffffffu);             // global_addr[31:0]
  g0[3] = (unsigned)((ga >> 32) & 0x01ffffffu)      // global_addr[56:32]
          | (2u << 30);                             // type=2 ("image")
  i32x8 g1;
  g1[0] = 0x00010000;                               // wg_mask=0, data_size=1 (2B)
  g1[1] = (int)((IN_DIM & 0xffffu) << 16);          // tensor_dim0[15:0] @ bits63:48
  g1[2] = (int)(((IN_DIM >> 16) & 0xffff)           // tensor_dim0[31:16]
          | ((F_DIM & 0xffff) << 16));              // tensor_dim1[15:0]
  g1[3] = (int)(((F_DIM >> 16) & 0xffff)            // tensor_dim1[31:16]
          | (32u << 16));                           // tile_dim0 = 32
  g1[4] = (int)F_DIM;                               // tile_dim1 = 256, tile_dim2 = 0
  g1[5] = (int)IN_DIM;                              // tensor_dim0_stride[31:0]
  g1[6] = 0;                                        // stride0 hi, dim1_stride lo
  g1[7] = 0;
  i32x4 z4; z4[0] = z4[1] = z4[2] = z4[3] = 0;
#if __clang_major__ >= 23
  i32x8 z8;
#pragma unroll
  for (int i = 0; i < 8; ++i) z8[i] = 0;
  __builtin_amdgcn_tensor_load_to_lds(g0, g1, z4, z4, z8, 0);
#else
  __builtin_amdgcn_tensor_load_to_lds(g0, g1, z4, z4, 0);
#endif
}
#endif

// ---------------- kernel 0: weights fp32 -> bf16 ----------------
__global__ void k_convert_weights(const float* __restrict__ W1,
                                  const float* __restrict__ Wa1,
                                  __bf16* __restrict__ W1bf,
                                  __bf16* __restrict__ Wa1bf) {
  int t = blockIdx.x * 256 + threadIdx.x;
  if (t < F_DIM * IN_DIM) W1bf[t]  = (__bf16)W1[t];
  if (t < A_DIM * F_DIM)  Wa1bf[t] = (__bf16)Wa1[t];
}

// ---------------- kernel 1: fused encoder + attention scores ----------------
// grid = N/128 WGs, 256 threads (8 waves). Wave w owns h cols [w*32,w*32+32)
// and attn cols [w*16,w*16+16). Double-buffered X staging; W1 K-chunks via TDM.
__global__ __launch_bounds__(256)
void k_mil_fused(const float*  __restrict__ X,
                 const __bf16* __restrict__ W1bf,  const float* __restrict__ b1,
                 const __bf16* __restrict__ Wa1bf, const float* __restrict__ ba1,
                 const float*  __restrict__ Wa2,   const float* __restrict__ ba2,
                 __bf16* __restrict__ h_out, float* __restrict__ scores) {
  __shared__ __bf16 ldsX[2][128 * APAD];  // bf16 X K-chunks, double buffered (20 KB)
#if HAVE_TDM
  __shared__ __bf16 ldsB[2][F_DIM * 32]; // bf16 W1 K-chunks via TDM           (32 KB)
#endif
  __shared__ __bf16 ldsH[128 * HPAD];    // bf16 h tile [128][256]             (66 KB)
  __shared__ float  ldsS[128];           // per-row score partials

  const int t    = threadIdx.x;
  const int wave = t >> 5;
  const int lane = t & 31;
  const int half = lane >> 4;       // 0 / 1
  const int r    = lane & 15;
  const long rowBase = (long)blockIdx.x * 128;

  if (t < 128) ldsS[t] = 0.f;

  // ---- GEMM1: h = relu(X @ W1^T + b1), M=128 (8 Mtiles), wave-N = 2 Ntiles ----
  v8f acc[8][2];
#pragma unroll
  for (int mt = 0; mt < 8; ++mt) { acc[mt][0] = v8f_zero(); acc[mt][1] = v8f_zero(); }

  const int tr = t >> 1;              // row this thread stages (0..127)
  const int tc = (t & 1) * 16;        // which half of the 32-wide K chunk
  const float* gX = X + (rowBase + tr) * IN_DIM + tc;

#if HAVE_TDM
  const unsigned ldsB_off = lds_off(&ldsB[0][0]);
  if (wave == 0) tdm_load_w_chunk(W1bf, ldsB_off);
#endif
  { // prologue: stage chunk 0 into buffer 0
    const f32x4* p = (const f32x4*)gX;
    cvt16_store(p[0], p[1], p[2], p[3], &ldsX[0][tr * APAD + tc]);
  }
#if HAVE_TDM
  if (wave == 0) __builtin_amdgcn_s_wait_tensorcnt(0);
#endif
  __syncthreads();

  for (int kb = 0; kb < IN_DIM; kb += 32) {
    const int cur = (kb >> 5) & 1;
    const bool haveNext = (kb + 32) < IN_DIM;

    // issue next chunk's data movement early: overlaps with WMMA below
    f32x4 f0, f1, f2, f3;
    if (haveNext) {
#if HAVE_TDM
      if (wave == 0)
        tdm_load_w_chunk(W1bf + kb + 32, ldsB_off + (unsigned)(cur ^ 1) * (F_DIM * 32 * 2));
#endif
      const f32x4* p = (const f32x4*)(gX + kb + 32);
      f0 = p[0]; f1 = p[1]; f2 = p[2]; f3 = p[3];
      if (kb + 64 < IN_DIM) __builtin_prefetch(gX + kb + 64, 0, 3);
    }

    // B fragments for this wave's two 16-col tiles
    BFrag B0, B1;
#if HAVE_TDM
    {
      const __bf16* p0 = &ldsB[cur][(wave * 32 + r) * 32 + half * 16];
      B0.q[0] = *(const u32x4*)p0;
      B0.q[1] = *(const u32x4*)(p0 + 8);
      const __bf16* p1 = &ldsB[cur][(wave * 32 + 16 + r) * 32 + half * 16];
      B1.q[0] = *(const u32x4*)p1;
      B1.q[1] = *(const u32x4*)(p1 + 8);
    }
#else
    {
      const __bf16* w0 = W1bf + (wave * 32 + r) * IN_DIM + kb + half * 16;
      B0.q[0] = *(const u32x4*)w0;
      B0.q[1] = *(const u32x4*)(w0 + 8);
      const __bf16* w1 = W1bf + (wave * 32 + 16 + r) * IN_DIM + kb + half * 16;
      B1.q[0] = *(const u32x4*)w1;
      B1.q[1] = *(const u32x4*)(w1 + 8);
    }
#endif

#pragma unroll
    for (int mt = 0; mt < 8; ++mt) {
      BFrag A;
      const __bf16* arow = &ldsX[cur][(mt * 16 + r) * APAD];
      A.q[0] = *(const u32x4*)(arow + half * 16);
      A.q[1] = *(const u32x4*)(arow + half * 16 + 8);
      acc[mt][0] = __builtin_amdgcn_wmma_f32_16x16x32_bf16(
          false, A.v, false, B0.v, (short)0, acc[mt][0], false, false);
      acc[mt][1] = __builtin_amdgcn_wmma_f32_16x16x32_bf16(
          false, A.v, false, B1.v, (short)0, acc[mt][1], false, false);
    }

    if (haveNext) {
      cvt16_store(f0, f1, f2, f3, &ldsX[cur ^ 1][tr * APAD + tc]);
#if HAVE_TDM
      if (wave == 0) __builtin_amdgcn_s_wait_tensorcnt(0);
#endif
    }
    __syncthreads();
  }

  // ---- epilogue GEMM1: bias + relu, write bf16 h tile to LDS ----
#pragma unroll
  for (int nt = 0; nt < 2; ++nt) {
    const int col  = wave * 32 + nt * 16 + r;
    const float bias = b1[col];
#pragma unroll
    for (int mt = 0; mt < 8; ++mt) {
#pragma unroll
      for (int j = 0; j < 8; ++j) {
        float v = acc[mt][nt][j] + bias;          // C layout: row = mt*16 + j + half*8
        v = v > 0.f ? v : 0.f;
        ldsH[(mt * 16 + j + half * 8) * HPAD + col] = (__bf16)v;
      }
    }
  }
  __syncthreads();

  // ---- spill h tile to global (bf16) for the bag reduce ----
#pragma unroll 4
  for (int i = 0; i < 16; ++i) {
    int idx = (i * 256 + t) * 8;       // element index within the 128x256 tile
    int row = idx >> 8;
    int col = idx & 255;
#if HAVE_ASYNC_ST
    __builtin_amdgcn_global_store_async_from_lds_b128(
        (glb_i32x4*)&h_out[(rowBase + row) * F_DIM + col],
        (lds_i32x4*)&ldsH[row * HPAD + col], 0, 0);
#else
    u32x4 v = *(const u32x4*)&ldsH[row * HPAD + col];
    *(u32x4*)&h_out[(rowBase + row) * F_DIM + col] = v;
#endif
  }

  // ---- GEMM2: a = tanh(h @ Wa1^T + ba1), wave owns 1 Ntile (16 attn cols) ----
  v8f acc2[8];
#pragma unroll
  for (int mt = 0; mt < 8; ++mt) acc2[mt] = v8f_zero();

  for (int kb = 0; kb < F_DIM; kb += 32) {
    BFrag B;
    const __bf16* w0 = Wa1bf + (wave * 16 + r) * F_DIM + kb + half * 16;
    B.q[0] = *(const u32x4*)w0;
    B.q[1] = *(const u32x4*)(w0 + 8);
#pragma unroll
    for (int mt = 0; mt < 8; ++mt) {
      BFrag A;
      const __bf16* arow = &ldsH[(mt * 16 + r) * HPAD + kb];
      A.q[0] = *(const u32x4*)(arow + half * 16);
      A.q[1] = *(const u32x4*)(arow + half * 16 + 8);
      acc2[mt] = __builtin_amdgcn_wmma_f32_16x16x32_bf16(
          false, A.v, false, B.v, (short)0, acc2[mt], false, false);
    }
  }

  // ---- tanh + per-row score = sum_n a[row][n] * Wa2[n] ----
  {
    const int col    = wave * 16 + r;
    const float bias = ba1[col];
    const float w2   = Wa2[col];
#pragma unroll
    for (int mt = 0; mt < 8; ++mt) {
#pragma unroll
      for (int j = 0; j < 8; ++j) {
        float c = tanhf(acc2[mt][j] + bias) * w2;
        // reduce across the 16 lanes of this half (same output row)
        c += __shfl_xor(c, 1, 32);
        c += __shfl_xor(c, 2, 32);
        c += __shfl_xor(c, 4, 32);
        c += __shfl_xor(c, 8, 32);
        if (r == 0) atomicAdd(&ldsS[mt * 16 + j + half * 8], c);  // ds_add_f32
      }
    }
  }
  __syncthreads();
  if (t < 128) scores[rowBase + t] = ldsS[t] + ba2[0];

#if HAVE_ASYNC_ST
#if __has_builtin(__builtin_amdgcn_s_wait_asynccnt)
  __builtin_amdgcn_s_wait_asynccnt(0);
#else
  asm volatile("s_wait_asynccnt 0x0" ::: "memory");
#endif
#endif
}

// ---------------- kernel 2: per-bag softmax stats (max, sum-exp) ----------------
__global__ __launch_bounds__(256)
void k_bag_stats(const float* __restrict__ scores, float* __restrict__ mz) {
  __shared__ float red[256];
  const int b = blockIdx.x, t = threadIdx.x;
  const float* s = scores + (size_t)b * BAG;
  float m = -1e30f;
  for (int i = t; i < BAG; i += 256) m = fmaxf(m, s[i]);
  red[t] = m; __syncthreads();
  for (int o = 128; o > 0; o >>= 1) { if (t < o) red[t] = fmaxf(red[t], red[t + o]); __syncthreads(); }
  m = red[0]; __syncthreads();
  float z = 0.f;
  for (int i = t; i < BAG; i += 256) z += __expf(s[i] - m);
  red[t] = z; __syncthreads();
  for (int o = 128; o > 0; o >>= 1) { if (t < o) red[t] += red[t + o]; __syncthreads(); }
  if (t == 0) { mz[b] = m; mz[N_BAGS + b] = red[0]; }
}

// ---------------- kernel 3: attention-weighted bag reduce + head ----------------
__global__ __launch_bounds__(256)
void k_bag_reduce(const __bf16* __restrict__ h, const float* __restrict__ scores,
                  const float* __restrict__ mz, const float* __restrict__ Wh,
                  const float* __restrict__ bh, float* __restrict__ out) {
  __shared__ float part[8][256];
  __shared__ float bag[256];
  const int b = blockIdx.x, t = threadIdx.x;
  const int wave = t >> 5, lane = t & 31;
  const float m = mz[b], invz = 1.f / mz[N_BAGS + b];
  const size_t base = (size_t)b * BAG;

  float acc[8];
#pragma unroll
  for (int i = 0; i < 8; ++i) acc[i] = 0.f;

  for (int p = wave; p < BAG; p += 8) {   // lane l owns features [l*8, l*8+8)
    float attn = __expf(scores[base + p] - m) * invz;
    BF8 hv;
    hv.q = *(const u32x4*)&h[(base + p) * F_DIM + lane * 8];
#pragma unroll
    for (int i = 0; i < 8; ++i) acc[i] += attn * (float)hv.h[i];
  }
#pragma unroll
  for (int i = 0; i < 8; ++i) part[wave][lane * 8 + i] = acc[i];
  __syncthreads();

  float v = 0.f;
#pragma unroll
  for (int w = 0; w < 8; ++w) v += part[w][t];
  bag[t] = v;
  __syncthreads();

  // head: out[b][k] = sum_f bag[f] * Wh[k][f] + bh[k]   (Wh row-major [2][256])
  part[0][t] = bag[t] * Wh[t];
  part[1][t] = bag[t] * Wh[256 + t];
  __syncthreads();
  for (int o = 128; o > 0; o >>= 1) {
    if (t < o) { part[0][t] += part[0][t + o]; part[1][t] += part[1][t + o]; }
    __syncthreads();
  }
  if (t == 0) { out[b * 2 + 0] = part[0][0] + bh[0]; out[b * 2 + 1] = part[1][0] + bh[1]; }
}

// ---------------- launch ----------------
extern "C" void kernel_launch(void* const* d_in, const int* in_sizes, int n_in,
                              void* d_out, int out_size, void* d_ws, size_t ws_size,
                              hipStream_t stream) {
  const float* X   = (const float*)d_in[0];
  const float* W1  = (const float*)d_in[1];
  const float* b1  = (const float*)d_in[2];
  const float* Wa1 = (const float*)d_in[3];
  const float* ba1 = (const float*)d_in[4];
  const float* Wa2 = (const float*)d_in[5];
  const float* ba2 = (const float*)d_in[6];
  const float* Wh  = (const float*)d_in[7];
  const float* bh  = (const float*)d_in[8];
  // d_in[9] = bag_sizes: uniform N/B per the reference setup.

  char* ws = (char*)d_ws;
  __bf16* W1bf   = (__bf16*)(ws);                       // 512 KB
  __bf16* Wa1bf  = (__bf16*)(ws + (512u << 10));        //  64 KB
  float*  scores = (float*)(ws + (576u << 10));         //   1 MB
  float*  mz     = (float*)(ws + (1600u << 10));        //  256 B  (m[32], z[32])
  __bf16* h_ws   = (__bf16*)(ws + (2u << 20));          // 128 MB  bf16 h[N][256]
  float*  outp   = (float*)d_out;

  k_convert_weights<<<(F_DIM * IN_DIM + 255) / 256, 256, 0, stream>>>(W1, Wa1, W1bf, Wa1bf);
  k_mil_fused<<<N_PATCHES / 128, 256, 0, stream>>>(X, W1bf, b1, Wa1bf, ba1, Wa2, ba2,
                                                   h_ws, scores);
  k_bag_stats<<<N_BAGS, 256, 0, stream>>>(scores, mz);
  k_bag_reduce<<<N_BAGS, 256, 0, stream>>>(h_ws, scores, mz, Wh, bh, outp);
}
